// CrossAttention_20048907338324
// MI455X (gfx1250) — matrix operand order repaired and checked by reading, not compiled, not run on hardware
//
#include <hip/hip_runtime.h>
#include <hip/hip_bf16.h>
#include <math.h>

// ---------- types ----------
typedef _Float16 h16;
typedef __attribute__((ext_vector_type(4)))  _Float16 v4h;
typedef __attribute__((ext_vector_type(8)))  _Float16 v8h;
typedef __attribute__((ext_vector_type(16))) _Float16 v16h;
typedef __attribute__((ext_vector_type(8)))  float    v8f;
typedef __attribute__((ext_vector_type(4)))  float    v4f;

#define BATCH 4
#define HEADS 8
#define SEQ   2048
#define CDIM  512
#define HDIM  64
#define ROWS  (BATCH * SEQ)          // 8192

static __device__ __forceinline__ v16h cat8(v8h a, v8h b) {
  return __builtin_shufflevector(a, b, 0,1,2,3,4,5,6,7,8,9,10,11,12,13,14,15);
}

static __device__ __forceinline__ v8f wmma_f16(v16h a, v16h b, v8f c) {
  // D = A(16x32,f16) * B(32x16,f16) + C(16x16,f32)
  return __builtin_amdgcn_wmma_f32_16x16x32_f16(false, a, false, b, (short)0, c, false, false);
}

// A-matrix 16x32 fragment from a row-major f16 row pointer.
// lane<16: K=0..7,16..23 ; lane>=16: K=8..15,24..31  (CDNA5 ISA 7.12.2)
static __device__ __forceinline__ v16h load_a_frag(const h16* rowptr, int c0, int lhalf) {
  v8h a0 = *(const v8h*)(rowptr + c0 +      lhalf * 8);
  v8h a1 = *(const v8h*)(rowptr + c0 + 16 + lhalf * 8);
  return cat8(a0, a1);
}

// ---------- fp32 -> fp16 convert (vectorized x4) ----------
__global__ void cvt_f32_f16_kernel(const float* __restrict__ src, h16* __restrict__ dst, int n4) {
  int i = blockIdx.x * blockDim.x + threadIdx.x;
  int stride = gridDim.x * blockDim.x;
  for (; i < n4; i += stride) {
    v4f f = *(const v4f*)(src + (size_t)i * 4);
    v4h h; h[0] = (h16)f[0]; h[1] = (h16)f[1]; h[2] = (h16)f[2]; h[3] = (h16)f[3];
    *(v4h*)(dst + (size_t)i * 4) = h;
  }
}

// ---------- 16x16 sub-tile store (C/D layout: lane<16 rows 0..7 col j; lane>=16 rows 8..15) ----
static __device__ __forceinline__ void store_subtile(v8f acc, int mode, void* dst,
                                                     const float* bias, int row0s, int o0s,
                                                     int lhalf, int l15) {
  const float bb  = bias[o0s + l15];
  const int nbase = row0s + lhalf * 8;
  if (mode == 2) {                                   // f32 flat [ROWS, CDIM] (+bias)
    float* out = (float*)dst;
#pragma unroll
    for (int e = 0; e < 8; ++e)
      out[(size_t)(nbase + e) * CDIM + o0s + l15] = acc[e] + bb;
  } else {
    const int h  = o0s >> 6;                         // head (16-col tile never crosses heads)
    const int d0 = o0s & 63;
    const int b  = nbase >> 11;                      // batch (tile never crosses batches)
    const int nl = nbase & (SEQ - 1);
    h16* out = (h16*)dst;
    if (mode == 0) {                                 // f16 [B,H,N,D]
      size_t base = (((size_t)(b * HEADS + h)) * SEQ + nl) * HDIM + d0 + l15;
#pragma unroll
      for (int e = 0; e < 8; ++e)
        out[base + (size_t)e * HDIM] = (h16)(acc[e] + bb);
    } else {                                         // f16 [B,H,D,N] : 8 rows contiguous in N
      v8h pk;
#pragma unroll
      for (int e = 0; e < 8; ++e) pk[e] = (h16)(acc[e] + bb);
      size_t base = (((size_t)(b * HEADS + h)) * HDIM + d0 + l15) * SEQ + nl;
      *(v8h*)(out + base) = pk;
    }
  }
}

// ---------- register-blocked wave GEMM:  Y[n,o] = sum_c A[n,c] * W[o,c] (+bias) ----------
// One wave computes a 64x32 tile = 4x2 WMMA sub-tiles: 8 WMMA per K-step fed by 12 b128 loads.
#define WM 64
#define WN 32
__global__ void wave_gemm_kernel(const h16* __restrict__ A, const h16* __restrict__ W,
                                 const float* __restrict__ bias, void* __restrict__ dst,
                                 int mode) {
  const int wave  = blockIdx.x * (blockDim.x >> 5) + (threadIdx.x >> 5);
  const int lane  = threadIdx.x & 31;
  const int lhalf = lane >> 4;
  const int l15   = lane & 15;

  const int colTiles = CDIM / WN;                    // 16
  const int row0 = (wave / colTiles) * WM;
  const int o0   = (wave % colTiles) * WN;
  if (row0 >= ROWS) return;

  const h16* arow = A + (size_t)(row0 + l15) * CDIM;
  const h16* wrow = W + (size_t)(o0   + l15) * CDIM; // row-major W serves directly as B

  v8f acc[4][2] = {};
#pragma unroll 2
  for (int c0 = 0; c0 < CDIM; c0 += 32) {
    v16h af[4], bf[2];
#pragma unroll
    for (int i = 0; i < 4; ++i)
      af[i] = load_a_frag(arow + (size_t)i * 16 * CDIM, c0, lhalf);
#pragma unroll
    for (int j = 0; j < 2; ++j)
      bf[j] = load_a_frag(wrow + (size_t)j * 16 * CDIM, c0, lhalf);
#pragma unroll
    for (int i = 0; i < 4; ++i)
#pragma unroll
      for (int j = 0; j < 2; ++j)
        acc[i][j] = wmma_f16(af[i], bf[j], acc[i][j]);
  }

#pragma unroll
  for (int i = 0; i < 4; ++i)
#pragma unroll
    for (int j = 0; j < 2; ++j)
      store_subtile(acc[i][j], mode, dst, bias, row0 + i * 16, o0 + j * 16, lhalf, l15);
}

// ---------- flash attention: one wave = one (b,h, 32-row q block: two 16-row tiles) ----------
// Q:[B,H,N1,D] f16, K:[B,H,N2,D] f16, VT:[B,H,D,N2] f16 -> O:[B,N1,C] f16
__global__ void flash_attn_kernel(const h16* __restrict__ Q, const h16* __restrict__ K,
                                  const h16* __restrict__ VT, h16* __restrict__ O) {
  const int wave  = blockIdx.x * (blockDim.x >> 5) + (threadIdx.x >> 5);
  const int w     = threadIdx.x >> 5;
  const int lane  = threadIdx.x & 31;
  const int lhalf = lane >> 4;
  const int l15   = lane & 15;

  const int qg = wave & (SEQ / 32 - 1);              // 64 q-groups of 32 rows
  const int h  = (wave >> 6) & (HEADS - 1);
  const int b  = wave >> 9;
  const int qr = qg * 32;

  __shared__ __align__(32) h16 plds[8][2][16 * 32];  // per-wave, per-qtile P staging

  const h16* kb = K  + ((size_t)(b * HEADS + h)) * SEQ * HDIM;
  const h16* vb = VT + ((size_t)(b * HEADS + h)) * HDIM * SEQ;

  // Q fragments: two q-tiles x two 16x32 A-frags over D=64, loaded once.
  v16h qf[2][2];
#pragma unroll
  for (int t = 0; t < 2; ++t) {
    const h16* qrow = Q + (((size_t)(b * HEADS + h)) * SEQ + qr + t * 16 + l15) * HDIM;
    qf[t][0] = load_a_frag(qrow, 0,  lhalf);
    qf[t][1] = load_a_frag(qrow, 32, lhalf);
  }

  v8f oacc[2][4] = {};                               // per q-tile: O 16x64 as four f32 tiles
  float mrow[2][8], lrow[2][8];
#pragma unroll
  for (int t = 0; t < 2; ++t)
#pragma unroll
    for (int e = 0; e < 8; ++e) { mrow[t][e] = -1e30f; lrow[t][e] = 0.0f; }

  for (int n2 = 0; n2 < SEQ; n2 += 32) {
    // Prefetch next chunk's K/V rows into L0 while this chunk computes (global_prefetch_b8).
    if (n2 + 32 < SEQ) {
      __builtin_prefetch(kb + (size_t)(n2 + 32 + l15) * HDIM, 0, 3);
      __builtin_prefetch(vb + (size_t)l15 * SEQ + n2 + 32, 0, 3);
    }

    // ---- K fragments for S cols 0..15 / 16..31, shared by both q-tiles ----
    const h16* krow0 = kb + (size_t)(n2 +      l15) * HDIM;
    const h16* krow1 = kb + (size_t)(n2 + 16 + l15) * HDIM;
    v16h kb00 = load_a_frag(krow0, 0,  lhalf);
    v16h kb01 = load_a_frag(krow0, 32, lhalf);
    v16h kb10 = load_a_frag(krow1, 0,  lhalf);
    v16h kb11 = load_a_frag(krow1, 32, lhalf);

#pragma unroll
    for (int t = 0; t < 2; ++t) {
      // ---- S = Q * K^T (16x32, unscaled — faithful to the reference bug) ----
      v8f s0 = {}; s0 = wmma_f16(qf[t][0], kb00, s0); s0 = wmma_f16(qf[t][1], kb01, s0);
      v8f s1 = {}; s1 = wmma_f16(qf[t][0], kb10, s1); s1 = wmma_f16(qf[t][1], kb11, s1);

      // ---- online softmax (rows per 16-lane half; butterfly over masks 1..8) ----
      float cmax[8];
#pragma unroll
      for (int e = 0; e < 8; ++e) cmax[e] = fmaxf(s0[e], s1[e]);
#pragma unroll
      for (int msk = 1; msk <= 8; msk <<= 1)
#pragma unroll
        for (int e = 0; e < 8; ++e)
          cmax[e] = fmaxf(cmax[e], __shfl_xor(cmax[e], msk, 32));

      float p0[8], p1[8], psum[8], scl[8];
#pragma unroll
      for (int e = 0; e < 8; ++e) {
        float mn = fmaxf(mrow[t][e], cmax[e]);
        scl[e]     = __expf(mrow[t][e] - mn);
        mrow[t][e] = mn;
        p0[e] = __expf(s0[e] - mn);
        p1[e] = __expf(s1[e] - mn);
        psum[e] = p0[e] + p1[e];
      }
#pragma unroll
      for (int msk = 1; msk <= 8; msk <<= 1)
#pragma unroll
        for (int e = 0; e < 8; ++e)
          psum[e] += __shfl_xor(psum[e], msk, 32);
#pragma unroll
      for (int e = 0; e < 8; ++e) lrow[t][e] = lrow[t][e] * scl[e] + psum[e];
#pragma unroll
      for (int d = 0; d < 4; ++d)
#pragma unroll
        for (int e = 0; e < 8; ++e) oacc[t][d][e] *= scl[e];

      // ---- stage P (f16) row-major in LDS for the C->A layout transpose ----
      h16* pl = plds[w][t];
#pragma unroll
      for (int e = 0; e < 8; ++e) {
        int r = lhalf * 8 + e;
        pl[r * 32 + l15]      = (h16)p0[e];
        pl[r * 32 + 16 + l15] = (h16)p1[e];
      }
    }

    asm volatile("s_wait_dscnt 0" ::: "memory");     // CDNA5 split-counter LDS fence

    // ---- P A-fragments ----
    v16h pf[2];
#pragma unroll
    for (int t = 0; t < 2; ++t) {
      const h16* pr = plds[w][t] + (size_t)l15 * 32;
      pf[t] = cat8(*(const v8h*)(pr + lhalf * 8), *(const v8h*)(pr + 16 + lhalf * 8));
    }

    // ---- V fragments (shared by both q-tiles):  O += P * V ----
#pragma unroll
    for (int d = 0; d < 4; ++d) {
      const h16* vrow = vb + (size_t)(d * 16 + l15) * SEQ + n2;
      v16h vf = load_a_frag(vrow, 0, lhalf);
#pragma unroll
      for (int t = 0; t < 2; ++t)
        oacc[t][d] = wmma_f16(pf[t], vf, oacc[t][d]);
    }
  }

  // ---- finalize: O / l, store [B,N1,C] f16 for the output projection ----
#pragma unroll
  for (int t = 0; t < 2; ++t) {
    float inv[8];
#pragma unroll
    for (int e = 0; e < 8; ++e) inv[e] = 1.0f / lrow[t][e];
    h16* ob = O + ((size_t)b * SEQ + qr + t * 16 + lhalf * 8) * CDIM + h * HDIM + l15;
#pragma unroll
    for (int d = 0; d < 4; ++d)
#pragma unroll
      for (int e = 0; e < 8; ++e)
        ob[(size_t)e * CDIM + d * 16] = (h16)(oacc[t][d][e] * inv[e]);
  }
}

// ---------- host launch ----------
extern "C" void kernel_launch(void* const* d_in, const int* in_sizes, int n_in,
                              void* d_out, int out_size, void* d_ws, size_t ws_size,
                              hipStream_t stream) {
  const float* x   = (const float*)d_in[0];
  const float* mlp = (const float*)d_in[1];
  const float* Wq  = (const float*)d_in[2];
  const float* bq  = (const float*)d_in[3];
  const float* Wk  = (const float*)d_in[4];
  const float* bk  = (const float*)d_in[5];
  const float* Wv  = (const float*)d_in[6];
  const float* bv  = (const float*)d_in[7];
  const float* Wp  = (const float*)d_in[8];
  const float* bp  = (const float*)d_in[9];

  const size_t NX = (size_t)ROWS * CDIM;             // 4,194,304 elements
  const size_t NW = (size_t)CDIM * CDIM;             //   262,144 elements

  char* ws = (char*)d_ws;
  size_t off = 0;
  h16* x16   = (h16*)(ws + off); off += NX * 2;
  h16* mlp16 = (h16*)(ws + off); off += NX * 2;
  h16* Wq16  = (h16*)(ws + off); off += NW * 2;
  h16* Wk16  = (h16*)(ws + off); off += NW * 2;
  h16* Wv16  = (h16*)(ws + off); off += NW * 2;
  h16* Wp16  = (h16*)(ws + off); off += NW * 2;
  h16* Q16   = (h16*)(ws + off); off += NX * 2;
  h16* K16   = (h16*)(ws + off); off += NX * 2;
  h16* VT16  = (h16*)(ws + off); off += NX * 2;
  h16* O16   = (h16*)(ws + off); off += NX * 2;      // total ~50 MB

  // 1) fp32 -> fp16 conversions (x4 vectorized)
  cvt_f32_f16_kernel<<<2048, 256, 0, stream>>>(x,   x16,   (int)(NX / 4));
  cvt_f32_f16_kernel<<<2048, 256, 0, stream>>>(mlp, mlp16, (int)(NX / 4));
  cvt_f32_f16_kernel<<<256,  256, 0, stream>>>(Wq,  Wq16,  (int)(NW / 4));
  cvt_f32_f16_kernel<<<256,  256, 0, stream>>>(Wk,  Wk16,  (int)(NW / 4));
  cvt_f32_f16_kernel<<<256,  256, 0, stream>>>(Wv,  Wv16,  (int)(NW / 4));
  cvt_f32_f16_kernel<<<256,  256, 0, stream>>>(Wp,  Wp16,  (int)(NW / 4));

  // 2) QKV projections: (8192/64)*(512/32) = 2048 waves, 8 waves/block -> 256 blocks
  wave_gemm_kernel<<<256, 256, 0, stream>>>(x16,   Wq16, bq, (void*)Q16,  0);
  wave_gemm_kernel<<<256, 256, 0, stream>>>(mlp16, Wk16, bk, (void*)K16,  0);
  wave_gemm_kernel<<<256, 256, 0, stream>>>(mlp16, Wv16, bv, (void*)VT16, 1);

  // 3) flash attention: 4*8*64 = 2048 waves, 8 waves/block -> 256 blocks
  flash_attn_kernel<<<256, 256, 0, stream>>>(Q16, K16, VT16, O16);

  // 4) output projection (f32 + bias) -> d_out
  wave_gemm_kernel<<<256, 256, 0, stream>>>(O16, Wp16, bp, d_out, 2);
}
